// CRF_52312701665883
// MI455X (gfx1250) — matrix operand order repaired and checked by reading, not compiled, not run on hardware
//
#include <hip/hip_runtime.h>
#include <hip/hip_fp16.h>

#define T_LEN 512
#define B_LEN 512
#define K_LEN 128
#define NT    8      // number of 16-wide N tiles (K_LEN/16)
#define ASTR  136    // padded LDS row stride in halfs (272B = 17*16B: aligned, conflict-light)

typedef __attribute__((ext_vector_type(16))) _Float16 v16h;
typedef __attribute__((ext_vector_type(8)))  _Float16 v8h;
typedef __attribute__((ext_vector_type(8)))  float    v8f;

// Wave-local LDS fence: each workgroup is a single wave32, so cross-lane LDS
// visibility only needs program order + DScnt==0. Crucially this does NOT
// drain LOADcnt, so prefetched global emission loads stay in flight across
// the fence and hide under the next step's WMMA sequence (a full __syncthreads
// lowers to s_wait_loadcnt_dscnt 0x0 and stalls the serial chain on HBM).
__device__ __forceinline__ void lds_fence_wave() {
  asm volatile("s_wait_dscnt 0x0" ::: "memory");
}

// Row reductions across the 16 lanes that hold one matrix row (xor masks 1..8
// stay inside each 16-lane half of the wave32, matching the WMMA C/D layout
// where lanes 0-15 hold rows v and lanes 16-31 hold rows v+8).
__device__ __forceinline__ float rowmax16(float m) {
  m = fmaxf(m, __shfl_xor(m, 1, 32));
  m = fmaxf(m, __shfl_xor(m, 2, 32));
  m = fmaxf(m, __shfl_xor(m, 4, 32));
  m = fmaxf(m, __shfl_xor(m, 8, 32));
  return m;
}
__device__ __forceinline__ float rowsum16(float s) {
  s += __shfl_xor(s, 1, 32);
  s += __shfl_xor(s, 2, 32);
  s += __shfl_xor(s, 4, 32);
  s += __shfl_xor(s, 8, 32);
  return s;
}

// ---------------------------------------------------------------------------
// Kernel 1: numerator (gather chain along the tagged path) + zero the output.
// ---------------------------------------------------------------------------
__global__ void crf_numerator(const float* __restrict__ em,
                              const int*   __restrict__ tags,
                              const int*   __restrict__ mask,
                              const float* __restrict__ startT,
                              const float* __restrict__ endT,
                              const float* __restrict__ trans,
                              float* __restrict__ num,
                              float* __restrict__ out)
{
  int b = blockIdx.x * blockDim.x + threadIdx.x;
  if (b == 0) out[0] = 0.0f;           // re-zero scalar output every call
  if (b >= B_LEN) return;
  int last  = tags[b];                 // t = 0
  float acc = startT[last] + em[(long)b * K_LEN + last];
  for (int t = 1; t < T_LEN; ++t) {
    if (mask[t * B_LEN + b]) {
      int cur = tags[t * B_LEN + b];
      acc += trans[last * K_LEN + cur] + em[((long)t * B_LEN + b) * K_LEN + cur];
      last = cur;
    }
  }
  acc += endT[last];
  num[b] = acc;
}

// ---------------------------------------------------------------------------
// Kernel 2: forward algorithm. One wave32 per block owns 16 batch rows.
// Log-domain recursion rewritten as f16 GEMM with f32 accumulation:
//   alpha[b,:] = R_b + y[b,:]   (y <= 0, exp(y) stored as f16 in LDS)
//   S = exp(y) @ exp(trans)     -> 32x v_wmma_f32_16x16x32_f16 per step
//   z = em + log(S); M = rowmax(z); y' = z - M; R += M   (masked blend)
// Emission loads for step t+1 are issued right after step t's combine, so
// they overlap the LDS fence + step t+1's entire WMMA sequence.
// ---------------------------------------------------------------------------
__global__ __launch_bounds__(32) void crf_forward(
    const float* __restrict__ em,   const int* __restrict__ mask,
    const float* __restrict__ startT, const float* __restrict__ endT,
    const float* __restrict__ trans, const float* __restrict__ num,
    float* __restrict__ out)
{
  __shared__ __align__(16) _Float16 sBT[K_LEN * ASTR]; // exp(trans)^T : sBT[n*ASTR+k]
  __shared__ __align__(16) _Float16 sA [16    * ASTR]; // exp(y)       : sA[r*ASTR+k]

  const int lane = threadIdx.x;
  const int ln   = lane & 15;
  const int h    = lane >> 4;          // which half of the wave (rows +8, K +16)
  const int base = blockIdx.x * 16;    // batch base for this wave

  // One-time: exp(transitions), transposed so B-fragments are contiguous along K.
  for (int idx = lane; idx < K_LEN * K_LEN; idx += 32) {
    int k = idx >> 7;
    int n = idx & (K_LEN - 1);
    sBT[n * ASTR + k] = (_Float16)__expf(trans[idx]);   // trans[k][n] -> sBT[n][k]
  }

  // t = 0: alpha0 = start + emissions[0]
  float y[8][NT];
  float R[8];
  {
    float sT[NT];
#pragma unroll
    for (int ti = 0; ti < NT; ++ti) sT[ti] = startT[ln + 16 * ti];
    const float* em0 = em + (long)base * K_LEN;
#pragma unroll
    for (int v = 0; v < 8; ++v) {
      float m = -1e30f;
#pragma unroll
      for (int ti = 0; ti < NT; ++ti) {
        float z = sT[ti] + em0[(v + 8 * h) * K_LEN + ln + 16 * ti];
        y[v][ti] = z;
        m = fmaxf(m, z);
      }
      m = rowmax16(m);
      R[v] = m;
#pragma unroll
      for (int ti = 0; ti < NT; ++ti) y[v][ti] -= m;
    }
  }
#pragma unroll
  for (int v = 0; v < 8; ++v)
#pragma unroll
    for (int ti = 0; ti < NT; ++ti)
      sA[(v + 8 * h) * ASTR + ln + 16 * ti] = (_Float16)__expf(y[v][ti]);

  // Software-pipelined emission/mask prefetch registers.
  float emr[8][NT];
  int   mv[8];
  auto load_step = [&](int t) {
    const float* emt = em + ((long)t * B_LEN + base) * K_LEN;
#pragma unroll
    for (int v = 0; v < 8; ++v)
#pragma unroll
      for (int ti = 0; ti < NT; ++ti)
        emr[v][ti] = emt[(v + 8 * h) * K_LEN + ln + 16 * ti];
#pragma unroll
    for (int v = 0; v < 8; ++v) mv[v] = mask[t * B_LEN + base + v + 8 * h];
  };
  load_step(1);   // in flight across the first fence + first GEMM

  // ---- forward scan over T (serial chain) ----
  for (int t = 1; t < T_LEN; ++t) {
    lds_fence_wave();  // previous step's sA stores -> this step's A-fragment loads

    v8f acc[NT] = {};
#pragma unroll
    for (int kb = 0; kb < K_LEN; kb += 32) {
      // A fragment (16x32 f16): per-lane two ds_load_b128 of 8 contiguous halfs
      const v8h a0 = *(const v8h*)&sA[ln * ASTR + kb + 16 * h];
      const v8h a1 = *(const v8h*)&sA[ln * ASTR + kb + 16 * h + 8];
      const v16h A = __builtin_shufflevector(a0, a1,
          0,1,2,3,4,5,6,7,8,9,10,11,12,13,14,15);
#pragma unroll
      for (int ti = 0; ti < NT; ++ti) {
        // B fragment (32x16 f16): contiguous along K because sBT is transposed
        const _Float16* bp = &sBT[(ln + 16 * ti) * ASTR + kb + 16 * h];
        const v8h b0 = *(const v8h*)bp;
        const v8h b1 = *(const v8h*)(bp + 8);
        const v16h Bm = __builtin_shufflevector(b0, b1,
            0,1,2,3,4,5,6,7,8,9,10,11,12,13,14,15);
        acc[ti] = __builtin_amdgcn_wmma_f32_16x16x32_f16(
            false, A, false, Bm, (short)0, acc[ti], false, false);
      }
    }

    // Combine in the native C/D layout: z = em + log(S); masked rescale.
    // (compiler inserts the loadcnt wait here, after the GEMM hid the latency)
#pragma unroll
    for (int v = 0; v < 8; ++v) {
      float zt[NT];
      float m = -1e30f;
#pragma unroll
      for (int ti = 0; ti < NT; ++ti) {
        float z = emr[v][ti] + __logf(acc[ti][v]);   // S >= ~0.9 always; log safe
        zt[ti] = z;
        m = fmaxf(m, z);
      }
      m = rowmax16(m);
      const bool upd = (mv[v] != 0);
      if (upd) R[v] += m;                             // uniform per row -> cndmask
#pragma unroll
      for (int ti = 0; ti < NT; ++ti) y[v][ti] = upd ? (zt[ti] - m) : y[v][ti];
    }

    // Issue next step's emission/mask loads now: they overlap the fence below,
    // the sA stores, and the entirety of step t+1's WMMA sequence.
    if (t + 1 < T_LEN) load_step(t + 1);

    lds_fence_wave();  // A-fragment reads complete before overwriting sA
#pragma unroll
    for (int v = 0; v < 8; ++v)
#pragma unroll
      for (int ti = 0; ti < NT; ++ti)
        sA[(v + 8 * h) * ASTR + ln + 16 * ti] = (_Float16)__expf(y[v][ti]);
  }

  // ---- denominator + log-likelihood contribution ----
  float eT[NT];
#pragma unroll
  for (int ti = 0; ti < NT; ++ti) eT[ti] = endT[ln + 16 * ti];
  float part = 0.0f;
#pragma unroll
  for (int v = 0; v < 8; ++v) {
    float w[NT];
    float m = -1e30f;
#pragma unroll
    for (int ti = 0; ti < NT; ++ti) {
      w[ti] = y[v][ti] + eT[ti];
      m = fmaxf(m, w[ti]);
    }
    m = rowmax16(m);
    float s = 0.0f;
#pragma unroll
    for (int ti = 0; ti < NT; ++ti) s += __expf(w[ti] - m);
    s = rowsum16(s);
    float den = R[v] + m + __logf(s);
    part += num[base + v + 8 * h] - den;    // numerator - denominator per row
  }
  if (ln == 0) atomicAdd(out, part);        // lanes 0 and 16: 8 rows each
}

// ---------------------------------------------------------------------------
extern "C" void kernel_launch(void* const* d_in, const int* in_sizes, int n_in,
                              void* d_out, int out_size, void* d_ws, size_t ws_size,
                              hipStream_t stream) {
  (void)in_sizes; (void)n_in; (void)out_size; (void)ws_size;
  const float* em     = (const float*)d_in[0];
  const int*   tags   = (const int*)  d_in[1];
  const int*   mask   = (const int*)  d_in[2];
  const float* startT = (const float*)d_in[3];
  const float* endT   = (const float*)d_in[4];
  const float* trans  = (const float*)d_in[5];
  float* out = (float*)d_out;
  float* num = (float*)d_ws;                 // 512 floats of scratch

  crf_numerator<<<dim3(2), dim3(256), 0, stream>>>(em, tags, mask, startT, endT,
                                                   trans, num, out);
  crf_forward<<<dim3(B_LEN / 16), dim3(32), 0, stream>>>(em, mask, startT, endT,
                                                         trans, num, out);
}
